// Decoder_58531814310243
// MI455X (gfx1250) — matrix operand order repaired and checked by reading, not compiled, not run on hardware
//
#include <hip/hip_runtime.h>

typedef __attribute__((ext_vector_type(16))) _Float16      v16h;
typedef __attribute__((ext_vector_type(8)))  float         v8f;
typedef __attribute__((ext_vector_type(4)))  unsigned int  v4u;

union F16Frag { v4u u[2]; v16h h; };
union F16x8   { v4u v;    _Float16 h[8]; };

__device__ inline v16h ld_frag(const _Float16* p, int second_off_h) {
    F16Frag f;
    f.u[0] = *(const v4u*)(p);
    f.u[1] = *(const v4u*)(p + second_off_h);
    return f.h;
}

__device__ inline v8f wmma16(v16h a, v16h b, v8f c) {
    return __builtin_amdgcn_wmma_f32_16x16x32_f16(false, a, false, b,
                                                  (short)0, c, false, false);
}

// Stage 64x32 row-major A chunk into LDS (row stride 32). Fully unrolled:
// both global b128 loads issued before any LDS store (MLP pipelining).
__device__ inline void stage_A_tile(const _Float16* __restrict__ A,
                                    _Float16* As, int m0, int kk, int K, int tid)
{
    const int c0 = tid, c1 = tid + 128;
    const int r0 = c0 >> 2, s0 = c0 & 3;
    const int r1 = c1 >> 2, s1 = c1 & 3;
    v4u a0 = *(const v4u*)&A[(size_t)(m0 + r0) * K + kk + s0 * 8];
    v4u a1 = *(const v4u*)&A[(size_t)(m0 + r1) * K + kk + s1 * 8];
    *(v4u*)&As[r0 * 32 + s0 * 8] = a0;
    *(v4u*)&As[r1 * 32 + s1 * 8] = a1;
}

// Stage a 32-row chunk of row-major B[K,N] into N-major LDS tile Bs[n*32+k].
// Coalesced global b128 reads (8 threads span a 128B row segment); loads for
// both chunks hoisted ahead of the b16 LDS scatter stores.
__device__ inline void stage_B_tile(const _Float16* __restrict__ B,
                                    _Float16* Bs, int kk, int n0, int N, int tid)
{
    const int c0 = tid, c1 = tid + 128;
    const int r0 = c0 >> 3, s0 = c0 & 7;
    const int r1 = c1 >> 3, s1 = c1 & 7;
    F16x8 t0, t1;
    t0.v = *(const v4u*)&B[(size_t)(kk + r0) * N + n0 + s0 * 8];
    t1.v = *(const v4u*)&B[(size_t)(kk + r1) * N + n0 + s1 * 8];
    #pragma unroll
    for (int i = 0; i < 8; ++i) Bs[(s0 * 8 + i) * 32 + r0] = t0.h[i];
    #pragma unroll
    for (int i = 0; i < 8; ++i) Bs[(s1 * 8 + i) * 32 + r1] = t1.h[i];
}

// ---------------------------------------------------------------------------
// Generic WMMA GEMM: C[M,N] = scale * (A[M,K] @ B[K,N]) (+ res), A/B f16
// row-major. Block tile 64x64, 128 threads (4 waves), wave tile 16x64.
// ---------------------------------------------------------------------------
__launch_bounds__(128)
__global__ void gemm_wmma_f16(const _Float16* __restrict__ A,
                              const _Float16* __restrict__ B,
                              float* __restrict__ Cf,
                              _Float16* __restrict__ Ch,
                              const float* __restrict__ res,
                              float scale, int M, int N, int K)
{
    __shared__ _Float16 As[64 * 32];   // row-major, row stride 32
    __shared__ _Float16 Bs[64 * 32];   // N-major: Bs[n*32 + k]

    const int tid  = threadIdx.x;
    const int wave = tid >> 5;
    const int lane = tid & 31;
    const int lh   = lane >> 4;    // 0: lanes 0-15, 1: lanes 16-31
    const int l16  = lane & 15;
    const int m0   = blockIdx.y * 64;
    const int n0   = blockIdx.x * 64;
    const int wm   = wave * 16;

    v8f acc[4] = {};

    for (int kk = 0; kk < K; kk += 32) {
        stage_A_tile(A, As, m0, kk, K, tid);
        stage_B_tile(B, Bs, kk, n0, N, tid);
        if (kk + 32 < K)
            __builtin_prefetch(&A[(size_t)(m0 + (tid >> 1)) * K + kk + 32], 0, 1);
        __syncthreads();

        // A fragment: lanes 0-15 hold K0..7 / K16..23, lanes 16-31 K8..15 / K24..31
        v16h af = ld_frag(&As[(wm + l16) * 32 + lh * 8], 16);
        #pragma unroll
        for (int j = 0; j < 4; ++j) {
            // B fragment (N-major): lanes 0-15 K0..7 / K16..23, lanes 16-31 K8..15 / K24..31
            v16h bf = ld_frag(&Bs[(16 * j + l16) * 32 + lh * 8], 16);
            acc[j] = wmma16(af, bf, acc[j]);
        }
        __syncthreads();
    }

    #pragma unroll
    for (int j = 0; j < 4; ++j)
        #pragma unroll
        for (int r = 0; r < 8; ++r) {
            int row = m0 + wm + r + 8 * lh;
            int col = n0 + 16 * j + l16;
            size_t idx = (size_t)row * N + col;
            float v = acc[j][r] * scale;
            if (res) v += res[idx];
            if (Cf) Cf[idx] = v;
            if (Ch) Ch[idx] = (_Float16)v;
        }
}

// ---------------------------------------------------------------------------
// Fused SwiGLU GEMM: C[M,N] = f16( silu(A@B1) * (A@B2) ).
// Shares the A tile across both matmuls; avoids 2x f32 [M,N] intermediates.
// ---------------------------------------------------------------------------
__launch_bounds__(128)
__global__ void gemm_swiglu_wmma(const _Float16* __restrict__ A,
                                 const _Float16* __restrict__ B1,
                                 const _Float16* __restrict__ B2,
                                 _Float16* __restrict__ Ch,
                                 int M, int N, int K)
{
    __shared__ _Float16 As [64 * 32];
    __shared__ _Float16 Bs1[64 * 32];
    __shared__ _Float16 Bs2[64 * 32];

    const int tid  = threadIdx.x;
    const int wave = tid >> 5;
    const int lane = tid & 31;
    const int lh   = lane >> 4;
    const int l16  = lane & 15;
    const int m0   = blockIdx.y * 64;
    const int n0   = blockIdx.x * 64;
    const int wm   = wave * 16;

    v8f acc1[4] = {}, acc2[4] = {};

    for (int kk = 0; kk < K; kk += 32) {
        stage_A_tile(A, As, m0, kk, K, tid);
        stage_B_tile(B1, Bs1, kk, n0, N, tid);
        stage_B_tile(B2, Bs2, kk, n0, N, tid);
        __syncthreads();

        v16h af = ld_frag(&As[(wm + l16) * 32 + lh * 8], 16);
        #pragma unroll
        for (int j = 0; j < 4; ++j) {
            v16h b1 = ld_frag(&Bs1[(16 * j + l16) * 32 + lh * 8], 16);
            v16h b2 = ld_frag(&Bs2[(16 * j + l16) * 32 + lh * 8], 16);
            acc1[j] = wmma16(af, b1, acc1[j]);
            acc2[j] = wmma16(af, b2, acc2[j]);
        }
        __syncthreads();
    }

    #pragma unroll
    for (int j = 0; j < 4; ++j)
        #pragma unroll
        for (int r = 0; r < 8; ++r) {
            int row = m0 + wm + r + 8 * lh;
            int col = n0 + 16 * j + l16;
            float a = acc1[j][r];
            float v = (a / (1.f + __expf(-a))) * acc2[j][r];
            Ch[(size_t)row * N + col] = (_Float16)v;
        }
}

// ---------------------------------------------------------------------------
// Differential attention (flash-style, dual online softmax), hs = 64, H = 16.
// Q/K layout: [(b*T+t), h*128 + part*64 + d] f16 ; V: [(b*S+s), h*64 + d].
// Output: per-head RMS-normalized, *(1-lambda_init), f16 [(b*T+t), h*64+d].
// ---------------------------------------------------------------------------
__device__ inline void online_update(v8f sc[4], float m[8], float l[8],
                                     v8f o[4], _Float16* ps,
                                     int wm, int lh, int l16)
{
    #pragma unroll
    for (int r = 0; r < 8; ++r) {
        float rm = fmaxf(fmaxf(sc[0][r], sc[1][r]), fmaxf(sc[2][r], sc[3][r]));
        rm = fmaxf(rm, __shfl_xor(rm, 1, 32));
        rm = fmaxf(rm, __shfl_xor(rm, 2, 32));
        rm = fmaxf(rm, __shfl_xor(rm, 4, 32));
        rm = fmaxf(rm, __shfl_xor(rm, 8, 32));
        float mn = fmaxf(m[r], rm);
        float f  = __expf(m[r] - mn);
        m[r] = mn;
        float rs = 0.f;
        #pragma unroll
        for (int j = 0; j < 4; ++j) {
            float p = __expf(sc[j][r] - mn);
            rs += p;
            ps[(wm + r + 8 * lh) * 64 + 16 * j + l16] = (_Float16)p;
            o[j][r] *= f;
        }
        rs += __shfl_xor(rs, 1, 32);
        rs += __shfl_xor(rs, 2, 32);
        rs += __shfl_xor(rs, 4, 32);
        rs += __shfl_xor(rs, 8, 32);
        l[r] = l[r] * f + rs;
    }
}

__launch_bounds__(128)
__global__ void diff_attn_wmma(const _Float16* __restrict__ Q,
                               const _Float16* __restrict__ Km,
                               const _Float16* __restrict__ Vm,
                               const float* __restrict__ lq1,
                               const float* __restrict__ lk1,
                               const float* __restrict__ lq2,
                               const float* __restrict__ lk2,
                               _Float16* __restrict__ O,
                               int T, int S, int causal)
{
    const int H = 16, HS = 64;
    const int QS = H * 2 * HS;   // 2048
    const int VS = H * HS;       // 1024
    const int t0 = blockIdx.x * 64;
    const int h  = blockIdx.y;
    const int b  = blockIdx.z;
    const int tid  = threadIdx.x;
    const int wave = tid >> 5, lane = tid & 31;
    const int lh   = lane >> 4, l16 = lane & 15;
    const int wm   = wave * 16;

    __shared__ _Float16 q1s[64 * 64], q2s[64 * 64];
    __shared__ _Float16 k1s[64 * 64], k2s[64 * 64];
    __shared__ _Float16 vts[64 * 64];                 // transposed: vts[d][s]
    __shared__ _Float16 p1s[64 * 64], p2s[64 * 64];

    {   // stage q1/q2 tiles once: all 8 loads issued, then 8 LDS stores
        v4u t1[4], t2[4];
        #pragma unroll
        for (int u = 0; u < 4; ++u) {
            int c = tid + u * 128;
            int row = c >> 3, seg = c & 7;
            size_t base = (size_t)(b * T + t0 + row) * QS + h * 2 * HS;
            t1[u] = *(const v4u*)&Q[base + seg * 8];
            t2[u] = *(const v4u*)&Q[base + HS + seg * 8];
        }
        #pragma unroll
        for (int u = 0; u < 4; ++u) {
            int c = tid + u * 128;
            int row = c >> 3, seg = c & 7;
            *(v4u*)&q1s[row * 64 + seg * 8] = t1[u];
            *(v4u*)&q2s[row * 64 + seg * 8] = t2[u];
        }
    }

    float d1 = 0.f, d2 = 0.f;
    for (int d = 0; d < HS; ++d) {
        d1 += lq1[h * HS + d] * lk1[h * HS + d];
        d2 += lq2[h * HS + d] * lk2[h * HS + d];
    }
    const float lam = __expf(d1) - __expf(d2) + 0.8f;

    float m1[8], l1[8], m2[8], l2[8];
    #pragma unroll
    for (int r = 0; r < 8; ++r) { m1[r] = -1e30f; l1[r] = 0.f; m2[r] = -1e30f; l2[r] = 0.f; }
    v8f o1[4] = {}, o2[4] = {};

    const int sEnd = causal ? (t0 + 64) : S;
    for (int s0 = 0; s0 < sEnd; s0 += 64) {
        __syncthreads();
        {   // stage k1/k2 tiles: hoist all global loads ahead of LDS stores
            v4u t1[4], t2[4];
            #pragma unroll
            for (int u = 0; u < 4; ++u) {
                int c = tid + u * 128;
                int row = c >> 3, seg = c & 7;
                size_t base = (size_t)(b * S + s0 + row) * QS + h * 2 * HS;
                t1[u] = *(const v4u*)&Km[base + seg * 8];
                t2[u] = *(const v4u*)&Km[base + HS + seg * 8];
            }
            #pragma unroll
            for (int u = 0; u < 4; ++u) {
                int c = tid + u * 128;
                int row = c >> 3, seg = c & 7;
                *(v4u*)&k1s[row * 64 + seg * 8] = t1[u];
                *(v4u*)&k2s[row * 64 + seg * 8] = t2[u];
            }
        }
        {   // V transposed into LDS (coalesced global reads, b16 LDS scatter)
            int s = tid & 63, dh = tid >> 6;
            const v4u* vp4 = (const v4u*)&Vm[(size_t)(b * S + s0 + s) * VS + h * HS + dh * 32];
            F16x8 t[4];
            #pragma unroll
            for (int c = 0; c < 4; ++c) t[c].v = vp4[c];
            #pragma unroll
            for (int c = 0; c < 4; ++c)
                #pragma unroll
                for (int i = 0; i < 8; ++i)
                    vts[(dh * 32 + c * 8 + i) * 64 + s] = t[c].h[i];
        }
        __syncthreads();

        v8f sc1[4] = {}, sc2[4] = {};
        #pragma unroll
        for (int kc = 0; kc < 2; ++kc) {
            v16h a1 = ld_frag(&q1s[(wm + l16) * 64 + kc * 32 + lh * 8], 16);
            v16h a2 = ld_frag(&q2s[(wm + l16) * 64 + kc * 32 + lh * 8], 16);
            #pragma unroll
            for (int j = 0; j < 4; ++j) {
                v16h b1 = ld_frag(&k1s[(16 * j + l16) * 64 + kc * 32 + lh * 8], 16);
                v16h b2 = ld_frag(&k2s[(16 * j + l16) * 64 + kc * 32 + lh * 8], 16);
                sc1[j] = wmma16(a1, b1, sc1[j]);
                sc2[j] = wmma16(a2, b2, sc2[j]);
            }
        }
        const bool diag = causal && (s0 == t0);
        #pragma unroll
        for (int j = 0; j < 4; ++j)
            #pragma unroll
            for (int r = 0; r < 8; ++r) {
                float v1 = sc1[j][r] * 0.125f;
                float v2 = sc2[j][r] * 0.125f;
                if (diag) {
                    int t = t0 + wm + r + 8 * lh;
                    int s = s0 + 16 * j + l16;
                    if (s > t) { v1 = -1e9f; v2 = -1e9f; }
                }
                sc1[j][r] = v1; sc2[j][r] = v2;
            }

        online_update(sc1, m1, l1, o1, p1s, wm, lh, l16);
        online_update(sc2, m2, l2, o2, p2s, wm, lh, l16);
        __syncthreads();

        #pragma unroll
        for (int kc = 0; kc < 2; ++kc) {
            v16h pa1 = ld_frag(&p1s[(wm + l16) * 64 + kc * 32 + lh * 8], 16);
            v16h pa2 = ld_frag(&p2s[(wm + l16) * 64 + kc * 32 + lh * 8], 16);
            #pragma unroll
            for (int j = 0; j < 4; ++j) {
                v16h vf = ld_frag(&vts[(16 * j + l16) * 64 + kc * 32 + lh * 8], 16);
                o1[j] = wmma16(pa1, vf, o1[j]);
                o2[j] = wmma16(pa2, vf, o2[j]);
            }
        }
    }

    float out[4][8];
    #pragma unroll
    for (int r = 0; r < 8; ++r) {
        float i1 = 1.f / l1[r], i2 = 1.f / l2[r];
        float ss = 0.f;
        #pragma unroll
        for (int j = 0; j < 4; ++j) {
            float v = o1[j][r] * i1 - lam * (o2[j][r] * i2);
            out[j][r] = v;
            ss += v * v;
        }
        ss += __shfl_xor(ss, 1, 32);
        ss += __shfl_xor(ss, 2, 32);
        ss += __shfl_xor(ss, 4, 32);
        ss += __shfl_xor(ss, 8, 32);
        float sc = rsqrtf(ss * (1.f / 64.f) + 1e-6f) * 0.2f;
        #pragma unroll
        for (int j = 0; j < 4; ++j) out[j][r] *= sc;
    }
    #pragma unroll
    for (int j = 0; j < 4; ++j)
        #pragma unroll
        for (int r = 0; r < 8; ++r) {
            int t = t0 + wm + r + 8 * lh;
            O[(size_t)(b * T + t) * VS + h * HS + 16 * j + l16] = (_Float16)out[j][r];
        }
}

// ---------------------------------------------------------------------------
// Small elementwise / norm kernels
// ---------------------------------------------------------------------------
__launch_bounds__(256)
__global__ void rmsnorm_f32(const float* __restrict__ x, const float* __restrict__ g,
                            float* __restrict__ of, _Float16* __restrict__ oh, int D)
{
    __shared__ float red[256];
    const size_t row = blockIdx.x;
    const float* xr = x + row * D;
    float s = 0.f;
    for (int i = threadIdx.x; i < D; i += 256) { float v = xr[i]; s += v * v; }
    red[threadIdx.x] = s;
    __syncthreads();
    for (int st = 128; st > 0; st >>= 1) {
        if (threadIdx.x < st) red[threadIdx.x] += red[threadIdx.x + st];
        __syncthreads();
    }
    float inv = rsqrtf(red[0] / (float)D + 1e-6f);
    for (int i = threadIdx.x; i < D; i += 256) {
        float v = xr[i] * inv * g[i];
        if (of) of[row * D + i] = v;
        if (oh) oh[row * D + i] = (_Float16)v;
    }
}

__global__ void cvt_f32_to_f16(const float* __restrict__ x, _Float16* __restrict__ y, size_t n)
{
    size_t i = (size_t)blockIdx.x * blockDim.x + threadIdx.x;
    size_t stride = (size_t)gridDim.x * blockDim.x;
    for (; i < n; i += stride) y[i] = (_Float16)x[i];
}

// ---------------------------------------------------------------------------
// Orchestration
// ---------------------------------------------------------------------------
extern "C" void kernel_launch(void* const* d_in, const int* in_sizes, int n_in,
                              void* d_out, int out_size, void* d_ws, size_t ws_size,
                              hipStream_t stream)
{
    (void)in_sizes; (void)n_in; (void)out_size; (void)ws_size;
    const int B = 4, T = 1024, D = 1024, H = 16, HS = 64, F = 4096;
    const int M = B * T;            // 4096
    const int Nqk = H * 2 * HS;     // 2048
    const int Nv  = H * HS;         // 1024

    const float* x    = (const float*)d_in[0];
    const float* enc  = (const float*)d_in[1];
    const float* Wq_s = (const float*)d_in[2];
    const float* Wk_s = (const float*)d_in[3];
    const float* Wv_s = (const float*)d_in[4];
    const float* Wo_s = (const float*)d_in[5];
    const float* lq1s = (const float*)d_in[6];
    const float* lk1s = (const float*)d_in[7];
    const float* lq2s = (const float*)d_in[8];
    const float* lk2s = (const float*)d_in[9];
    const float* Wq_c = (const float*)d_in[10];
    const float* Wk_c = (const float*)d_in[11];
    const float* Wv_c = (const float*)d_in[12];
    const float* Wo_c = (const float*)d_in[13];
    const float* lq1c = (const float*)d_in[14];
    const float* lk1c = (const float*)d_in[15];
    const float* lq2c = (const float*)d_in[16];
    const float* lk2c = (const float*)d_in[17];
    const float* g    = (const float*)d_in[18];
    const float* W1   = (const float*)d_in[19];
    const float* W2   = (const float*)d_in[20];
    const float* W3   = (const float*)d_in[21];

    char* ws = (char*)d_ws;
    size_t off = 0;
    auto alloc = [&](size_t bytes) -> void* {
        off = (off + 255) & ~(size_t)255;
        void* p = ws + off;
        off += bytes;
        return p;
    };

    float*     h1f  = (float*)    alloc((size_t)M * D * 4);
    _Float16*  h1h  = (_Float16*) alloc((size_t)M * D * 2);
    _Float16*  ench = (_Float16*) alloc((size_t)M * D * 2);
    _Float16*  wqs  = (_Float16*) alloc((size_t)D * Nqk * 2);
    _Float16*  wks  = (_Float16*) alloc((size_t)D * Nqk * 2);
    _Float16*  wvs  = (_Float16*) alloc((size_t)D * Nv * 2);
    _Float16*  wos  = (_Float16*) alloc((size_t)Nv * D * 2);
    _Float16*  wqc  = (_Float16*) alloc((size_t)D * Nqk * 2);
    _Float16*  wkc  = (_Float16*) alloc((size_t)D * Nqk * 2);
    _Float16*  wvc  = (_Float16*) alloc((size_t)D * Nv * 2);
    _Float16*  woc  = (_Float16*) alloc((size_t)Nv * D * 2);
    _Float16*  w1h  = (_Float16*) alloc((size_t)D * F * 2);
    _Float16*  w2h  = (_Float16*) alloc((size_t)D * F * 2);
    _Float16*  w3h  = (_Float16*) alloc((size_t)F * D * 2);
    _Float16*  h2h  = (_Float16*) alloc((size_t)M * D * 2);
    float*     h4f  = (float*)    alloc((size_t)M * D * 4);
    _Float16*  rh   = (_Float16*) alloc((size_t)M * D * 2);
    _Float16*  attn = (_Float16*) alloc((size_t)M * Nv * 2);
    _Float16*  qb   = (_Float16*) alloc((size_t)M * Nqk * 2);
    _Float16*  kb   = (_Float16*) alloc((size_t)M * Nqk * 2);
    _Float16*  vb   = (_Float16*) alloc((size_t)M * Nv * 2);
    _Float16*  wglu = qb;  // alias: q/k region (33.5MB) free by SwiGLU time

    const dim3 cg(4096), cb(256);
    cvt_f32_to_f16<<<cg, cb, 0, stream>>>(enc,  ench, (size_t)M * D);
    cvt_f32_to_f16<<<cg, cb, 0, stream>>>(Wq_s, wqs, (size_t)D * Nqk);
    cvt_f32_to_f16<<<cg, cb, 0, stream>>>(Wk_s, wks, (size_t)D * Nqk);
    cvt_f32_to_f16<<<cg, cb, 0, stream>>>(Wv_s, wvs, (size_t)D * Nv);
    cvt_f32_to_f16<<<cg, cb, 0, stream>>>(Wo_s, wos, (size_t)Nv * D);
    cvt_f32_to_f16<<<cg, cb, 0, stream>>>(Wq_c, wqc, (size_t)D * Nqk);
    cvt_f32_to_f16<<<cg, cb, 0, stream>>>(Wk_c, wkc, (size_t)D * Nqk);
    cvt_f32_to_f16<<<cg, cb, 0, stream>>>(Wv_c, wvc, (size_t)D * Nv);
    cvt_f32_to_f16<<<cg, cb, 0, stream>>>(Wo_c, woc, (size_t)Nv * D);
    cvt_f32_to_f16<<<cg, cb, 0, stream>>>(W1, w1h, (size_t)D * F);
    cvt_f32_to_f16<<<cg, cb, 0, stream>>>(W2, w2h, (size_t)D * F);
    cvt_f32_to_f16<<<cg, cb, 0, stream>>>(W3, w3h, (size_t)F * D);

    rmsnorm_f32<<<M, 256, 0, stream>>>(x, g, h1f, h1h, D);

    auto ggrid = [](int N, int Mm) { return dim3(N / 64, Mm / 64); };

    // self-attention
    gemm_wmma_f16<<<ggrid(Nqk, M), 128, 0, stream>>>(h1h, wqs, nullptr, qb, nullptr, 1.f, M, Nqk, D);
    gemm_wmma_f16<<<ggrid(Nqk, M), 128, 0, stream>>>(h1h, wks, nullptr, kb, nullptr, 1.f, M, Nqk, D);
    gemm_wmma_f16<<<ggrid(Nv,  M), 128, 0, stream>>>(h1h, wvs, nullptr, vb, nullptr, 1.f, M, Nv,  D);
    diff_attn_wmma<<<dim3(T / 64, H, B), 128, 0, stream>>>(qb, kb, vb, lq1s, lk1s, lq2s, lk2s,
                                                           attn, T, T, 1);
    gemm_wmma_f16<<<ggrid(D, M), 128, 0, stream>>>(attn, wos, nullptr, h2h, h1f, 1.f, M, D, Nv);

    // cross-attention
    gemm_wmma_f16<<<ggrid(Nqk, M), 128, 0, stream>>>(h2h,  wqc, nullptr, qb, nullptr, 1.f, M, Nqk, D);
    gemm_wmma_f16<<<ggrid(Nqk, M), 128, 0, stream>>>(ench, wkc, nullptr, kb, nullptr, 1.f, M, Nqk, D);
    gemm_wmma_f16<<<ggrid(Nv,  M), 128, 0, stream>>>(ench, wvc, nullptr, vb, nullptr, 1.f, M, Nv,  D);
    diff_attn_wmma<<<dim3(T / 64, H, B), 128, 0, stream>>>(qb, kb, vb, lq1c, lk1c, lq2c, lk2c,
                                                           attn, T, T, 0);
    // h4 = 2 * (attn @ Wo_c)   (implements "h = h + h")
    gemm_wmma_f16<<<ggrid(D, M), 128, 0, stream>>>(attn, woc, h4f, nullptr, nullptr, 2.f, M, D, Nv);

    // out = swiglu(rmsnorm(h4)) @ W3 + h4  (fused SwiGLU GEMM)
    rmsnorm_f32<<<M, 256, 0, stream>>>(h4f, g, nullptr, rh, D);
    gemm_swiglu_wmma<<<ggrid(F, M), 128, 0, stream>>>(rh, w1h, w2h, wglu, M, F, D);
    gemm_wmma_f16<<<ggrid(D, M), 128, 0, stream>>>(wglu, w3h, (float*)d_out, nullptr, h4f,
                                                   1.f, M, D, F);
}
